// TransformerDecoderOnly_80762565034446
// MI455X (gfx1250) — compile-verified, matrix operand order checked by reading, not run on hardware
//
#include <hip/hip_runtime.h>
#include <cstdint>

typedef __bf16 bf16;
typedef __attribute__((ext_vector_type(16))) __bf16 v16bf;
typedef __attribute__((ext_vector_type(8)))  float  v8f;
typedef __attribute__((ext_vector_type(4)))  unsigned u32x4;
typedef __attribute__((ext_vector_type(4)))  int      i32x4;
typedef __attribute__((ext_vector_type(8)))  int      i32x8;

constexpr int kV = 32000, kD = 512, kS = 2048, kB = 2, kL = 6, kH = 8;
constexpr int kDh = kD / kH;   // 64
constexpr int kF  = 4 * kD;    // 2048
constexpr int kM  = kB * kS;   // 4096

static __device__ __forceinline__ bf16 f2bf(float f) {
  unsigned u = __builtin_bit_cast(unsigned, f);
  u += 0x7FFFu + ((u >> 16) & 1u);               // round-to-nearest-even
  unsigned short h = (unsigned short)(u >> 16);
  return __builtin_bit_cast(bf16, h);
}

union AF {
  v16bf v;
  unsigned u32[8];
  bf16 h[16];
};

// ---------------------------------------------------------------------------
// CDNA5 async global->LDS copy (16B per lane) + asynccnt fence
// ---------------------------------------------------------------------------
static __device__ __forceinline__ void async_ld_b128(void* lds, const void* g) {
  unsigned l = (unsigned)(size_t)lds;                     // LDS addr (low 32b)
  unsigned long long a = (unsigned long long)(size_t)g;   // 64-bit global addr
  asm volatile("global_load_async_to_lds_b128 %0, %1, off"
               :: "v"(l), "v"(a) : "memory");
}
static __device__ __forceinline__ void wait_async0() {
  asm volatile("s_wait_asynccnt 0" ::: "memory");
}

// ---------------------------------------------------------------------------
// CDNA5 Tensor Data Mover: DMA a 2D tile (rows x tile_k bf16 elems, row
// stride = stride0 elems) from global into LDS, inserting 4 DWORDs of pad
// every 32 DWORDs (128B row -> 144B LDS stride = 72 elems).
// Descriptor packing per CDNA5 ISA.
// ---------------------------------------------------------------------------
static __device__ __forceinline__ void tdm_load_tile(
    void* lds, const void* gptr, unsigned tile_k, unsigned rows,
    unsigned long long stride0) {
  unsigned long long ga = (unsigned long long)(size_t)gptr;
  u32x4 g0;
  g0[0] = 1u;                                            // count=1 (valid D#)
  g0[1] = (unsigned)(size_t)lds;                         // lds_addr
  g0[2] = (unsigned)(ga & 0xFFFFFFFFu);                  // global_addr[31:0]
  g0[3] = (unsigned)((ga >> 32) & 0x01FFFFFFu) | (2u << 30);  // [56:32] | type=2
  i32x8 g1;
  // workgroup_mask=0 | data_size=1 (2B) | pad_enable | pad_interval=4 (32DW)
  // | pad_amount=3 (4DW)
  g1[0] = (1 << 16) | (1 << 20) | (4 << 22) | (3 << 25);
  g1[1] = (int)((tile_k & 0xFFFFu) << 16);               // tensor_dim0[15:0]
  g1[2] = (int)((tile_k >> 16) | ((rows & 0xFFFFu) << 16));   // d0 hi | dim1 lo
  g1[3] = (int)((rows >> 16) | (tile_k << 16));          // dim1 hi | tile_dim0
  g1[4] = (int)(rows & 0xFFFFu);                         // tile_dim1 (dim2=0)
  g1[5] = (int)(stride0 & 0xFFFFFFFFu);                  // tensor_dim0_stride lo
  g1[6] = (int)((stride0 >> 32) & 0xFFFFu);              // stride hi (dim1_stride=0)
  g1[7] = 0;
  i32x4 z4 = {0, 0, 0, 0};
#if __has_include(<hip/amd_detail/amd_gfx1250_TDM.h>)
  i32x8 z8 = {0, 0, 0, 0, 0, 0, 0, 0};                   // 6-arg toolchain
  __builtin_amdgcn_tensor_load_to_lds(g0, g1, z4, z4, z8, 0);
#else
  __builtin_amdgcn_tensor_load_to_lds(g0, g1, z4, z4, 0); // ROCm 7.2 (5-arg)
#endif
}

// ---------------------------------------------------------------------------
// fp32 [K,N] -> bf16 transposed [N,K]  (tiled via LDS; one-time per launch)
// ---------------------------------------------------------------------------
__global__ __launch_bounds__(256)
void cvt_t_kernel(const float* __restrict__ src, bf16* __restrict__ dst,
                  int K, int N) {
  __shared__ bf16 t[32][33];
  const int n0 = blockIdx.x * 32, k0 = blockIdx.y * 32;
  const int tx = threadIdx.x & 31, ty = threadIdx.x >> 5;   // 32 x 8
#pragma unroll
  for (int i = 0; i < 4; ++i)
    t[ty + 8 * i][tx] = f2bf(src[(size_t)(k0 + ty + 8 * i) * N + n0 + tx]);
  __syncthreads();
#pragma unroll
  for (int i = 0; i < 4; ++i)
    dst[(size_t)(n0 + ty + 8 * i) * K + k0 + tx] = t[tx][ty + 8 * i];
}

// ---------------------------------------------------------------------------
// Embedding gather + positional encoding
// ---------------------------------------------------------------------------
__global__ __launch_bounds__(256)
void embed_kernel(const int* __restrict__ x, const float* __restrict__ emb,
                  const float* __restrict__ pe, float* __restrict__ h,
                  bf16* __restrict__ hbf) {
  int idx  = blockIdx.x * 256 + threadIdx.x;     // over kM*kD
  int row  = idx >> 9;
  int dcol = idx & (kD - 1);
  int s    = row & (kS - 1);
  float v = emb[(size_t)x[row] * kD + dcol] + pe[(size_t)s * kD + dcol];
  h[idx]   = v;
  hbf[idx] = f2bf(v);
}

// ---------------------------------------------------------------------------
// WMMA GEMM: C[M,N] = A[M,K](bf16) * W[K,N], weights given TRANSPOSED [N,K].
// Block tile 64x128, BK=64. Tiles are DMA'd into double-buffered LDS by the
// Tensor Data Mover (wave 0 issues the descriptors, TENSORcnt fences them).
// 8 waves of 32x32 (2x2 v_wmma_f32_16x16x32_bf16 per 32-K step); fragments
// are contiguous 16B LDS reads. EPI: 0=none,1=+bias,2=+bias+relu. OBF: bf16.
// ---------------------------------------------------------------------------
constexpr int BM = 64, BN = 128, BK = 64;
constexpr int LDK = BK + 8;    // 72 (matches TDM pad: 128B rows + 16B pad)

template <int EPI, bool OBF>
__global__ __launch_bounds__(256)
void gemm_bf16_kernel(const bf16* __restrict__ A, const bf16* __restrict__ Bt,
                      const float* __restrict__ bias, void* __restrict__ Cp,
                      int MM, int NN, int KK) {
  __shared__ __align__(16) bf16 As[2][BM * LDK];
  __shared__ __align__(16) bf16 Bs[2][BN * LDK];

  const int tid  = threadIdx.x;
  const int lane = tid & 31;
  const int half = lane >> 4;
  const int l16  = lane & 15;
  const int wave = tid >> 5;
  const int wm   = (wave >> 2) * 32;
  const int wn   = (wave & 3) * 32;
  const int bm   = blockIdx.y * BM;
  const int bn   = blockIdx.x * BN;

  v8f acc[2][2] = {};

  auto issue = [&](int buf, int k0) {
    if (wave == 0) {
      tdm_load_tile(&As[buf][0], &A[(size_t)bm * KK + k0], BK, BM,
                    (unsigned long long)KK);
      tdm_load_tile(&Bs[buf][0], &Bt[(size_t)bn * KK + k0], BK, BN,
                    (unsigned long long)KK);
    }
  };

  issue(0, 0);
  const int nk = KK / BK;
  for (int kt = 0; kt < nk; ++kt) {
    __builtin_amdgcn_s_wait_tensorcnt(0);    // wave0's DMAs complete
    __syncthreads();                         // tile kt visible to all waves
    if (kt + 1 < nk) issue((kt + 1) & 1, (kt + 1) * BK);
    const bf16* as = As[kt & 1];
    const bf16* bs = Bs[kt & 1];
#pragma unroll
    for (int kk = 0; kk < BK; kk += 32) {
      AF a[2], b[2];
#pragma unroll
      for (int i = 0; i < 2; ++i) {
        const bf16* p = &as[(wm + i * 16 + l16) * LDK + kk + half * 8];
        *(uint4*)&a[i].u32[0] = *(const uint4*)p;
        *(uint4*)&a[i].u32[4] = *(const uint4*)(p + 16);
      }
#pragma unroll
      for (int j = 0; j < 2; ++j) {
        const bf16* p = &bs[(wn + j * 16 + l16) * LDK + kk + half * 8];
        *(uint4*)&b[j].u32[0] = *(const uint4*)p;
        *(uint4*)&b[j].u32[4] = *(const uint4*)(p + 16);
      }
#pragma unroll
      for (int i = 0; i < 2; ++i)
#pragma unroll
        for (int j = 0; j < 2; ++j)
          acc[i][j] = __builtin_amdgcn_wmma_f32_16x16x32_bf16(
              false, a[i].v, false, b[j].v, (short)0, acc[i][j], false, false);
    }
  }

#pragma unroll
  for (int i = 0; i < 2; ++i)
#pragma unroll
    for (int j = 0; j < 2; ++j) {
      int n = bn + wn + j * 16 + l16;
      float bv = 0.f;
      if (EPI >= 1) bv = bias[n];
#pragma unroll
      for (int r = 0; r < 8; ++r) {
        int m = bm + wm + i * 16 + r + half * 8;
        float cvl = acc[i][j][r] + bv;
        if (EPI == 2) cvl = fmaxf(cvl, 0.f);
        size_t idx = (size_t)m * NN + n;
        if (OBF) ((bf16*)Cp)[idx] = f2bf(cvl);
        else     ((float*)Cp)[idx] = cvl;
      }
    }
}

// ---------------------------------------------------------------------------
// Flash attention: one block per (b, h, 64-query tile); 4 waves x 16 q-rows.
// Q and K tiles stream in via async LDS copies (K stays row-major [kv][dh]:
// that IS the fragment-friendly layout for Q*K^T). V is transposed on store.
// ---------------------------------------------------------------------------
__global__ __launch_bounds__(128)
void attn_kernel(const bf16* __restrict__ qkv, bf16* __restrict__ o) {
  const int qb   = blockIdx.x;
  const int bh   = blockIdx.y;
  const int b    = bh >> 3;
  const int hh   = bh & (kH - 1);
  const int tid  = threadIdx.x;
  const int lane = tid & 31;
  const int wave = tid >> 5;
  const int half = lane >> 4;
  const int l16  = lane & 15;
  constexpr int LDT = kDh + 8;           // 72
  constexpr int LDQ = 3 * kD;            // 1536

  __shared__ __align__(16) bf16 Qs [64 * LDT];   // [q][dh]
  __shared__ __align__(16) bf16 Ks [64 * LDT];   // [kv][dh]
  __shared__ __align__(16) bf16 VsT[64 * LDT];   // [dh][kv]
  __shared__ __align__(16) bf16 Ps [64 * LDT];   // [q][kv]

  const size_t rb = (size_t)b * kS;

#pragma unroll
  for (int c = 0; c < 4; ++c) {          // Q tile 64x64 via async copy
    int q = tid + c * 128;
    int r = q >> 3, col = (q & 7) << 3;
    async_ld_b128(&Qs[r * LDT + col],
                  &qkv[(rb + qb * 64 + r) * LDQ + hh * kDh + col]);
  }

  v8f acc_o[4] = {};
  float mrun[8], lrun[8];
#pragma unroll
  for (int r = 0; r < 8; ++r) { mrun[r] = -1e30f; lrun[r] = 0.f; }

  for (int j = 0; j <= qb; ++j) {        // causal: skip blocks above diagonal
    __syncthreads();                     // previous iteration's reads done
#pragma unroll
    for (int c = 0; c < 4; ++c) {        // K tile: async, natural layout
      int q = tid + c * 128;
      int r = q >> 3, col = (q & 7) << 3;
      async_ld_b128(&Ks[r * LDT + col],
                    &qkv[(rb + j * 64 + r) * LDQ + kD + hh * kDh + col]);
    }
#pragma unroll
    for (int c = 0; c < 4; ++c) {        // V tile: transpose on store
      int q = tid + c * 128;
      int r = q >> 3, col = (q & 7) << 3;
      uint4 vd = *(const uint4*)(&qkv[(rb + j * 64 + r) * LDQ + 2 * kD + hh * kDh + col]);
      const bf16* vp = (const bf16*)&vd;
#pragma unroll
      for (int e = 0; e < 8; ++e) VsT[(col + e) * LDT + r] = vp[e];
    }
    wait_async0();
    __syncthreads();

    // S = Q * K^T : each wave computes its 16x64 strip
    v8f sc[4] = {};
#pragma unroll
    for (int kk = 0; kk < kDh; kk += 32) {
      AF a;
      {
        const bf16* p = &Qs[(wave * 16 + l16) * LDT + kk + half * 8];
        *(uint4*)&a.u32[0] = *(const uint4*)p;
        *(uint4*)&a.u32[4] = *(const uint4*)(p + 16);
      }
#pragma unroll
      for (int t = 0; t < 4; ++t) {
        AF bfr;
        const bf16* p = &Ks[(t * 16 + l16) * LDT + kk + half * 8];
        *(uint4*)&bfr.u32[0] = *(const uint4*)p;
        *(uint4*)&bfr.u32[4] = *(const uint4*)(p + 16);
        sc[t] = __builtin_amdgcn_wmma_f32_16x16x32_bf16(
            false, a.v, false, bfr.v, (short)0, sc[t], false, false);
      }
    }

    // online softmax on the C layout (row across 16 lanes of one VGPR)
    const bool diag = (j == qb);
#pragma unroll
    for (int r = 0; r < 8; ++r) {
      int grow = qb * 64 + wave * 16 + r + half * 8;
      float sv[4];
      float rmax = -1e30f;
#pragma unroll
      for (int t = 0; t < 4; ++t) {
        float xv = sc[t][r] * 0.125f;          // 1/sqrt(Dh)
        int gcol = j * 64 + t * 16 + l16;
        if (diag && gcol > grow) xv = -1e30f;  // causal mask
        sv[t] = xv;
        rmax = fmaxf(rmax, xv);
      }
#pragma unroll
      for (int m = 1; m < 16; m <<= 1) rmax = fmaxf(rmax, __shfl_xor(rmax, m));
      float mnew  = fmaxf(mrun[r], rmax);
      float alpha = __expf(mrun[r] - mnew);
      mrun[r] = mnew;
      lrun[r] *= alpha;
#pragma unroll
      for (int t = 0; t < 4; ++t) acc_o[t][r] = acc_o[t][r] * alpha;
      float rs = 0.f;
#pragma unroll
      for (int t = 0; t < 4; ++t) {
        float p = __expf(sv[t] - mnew);
        rs += p;
        Ps[(wave * 16 + r + half * 8) * LDT + t * 16 + l16] = f2bf(p);
      }
#pragma unroll
      for (int m = 1; m < 16; m <<= 1) rs += __shfl_xor(rs, m);
      lrun[r] += rs;
    }
    // Ps strip is per-wave private; same-wave LDS ops are in-order

    // O += P * V
#pragma unroll
    for (int kk = 0; kk < 64; kk += 32) {
      AF a;
      {
        const bf16* p = &Ps[(wave * 16 + l16) * LDT + kk + half * 8];
        *(uint4*)&a.u32[0] = *(const uint4*)p;
        *(uint4*)&a.u32[4] = *(const uint4*)(p + 16);
      }
#pragma unroll
      for (int t = 0; t < 4; ++t) {
        AF bfr;
        const bf16* p = &VsT[(t * 16 + l16) * LDT + kk + half * 8];
        *(uint4*)&bfr.u32[0] = *(const uint4*)p;
        *(uint4*)&bfr.u32[4] = *(const uint4*)(p + 16);
        acc_o[t] = __builtin_amdgcn_wmma_f32_16x16x32_bf16(
            false, a.v, false, bfr.v, (short)0, acc_o[t], false, false);
      }
    }
  }

#pragma unroll
  for (int t = 0; t < 4; ++t)
#pragma unroll
    for (int r = 0; r < 8; ++r) {
      int grow = qb * 64 + wave * 16 + r + half * 8;
      float ov = acc_o[t][r] / lrun[r];
      o[(rb + grow) * (size_t)kD + hh * kDh + t * 16 + l16] = f2bf(ov);
    }
}

// ---------------------------------------------------------------------------
// Fused (residual +) LayerNorm. One block (256 thr) per row of 512.
// ---------------------------------------------------------------------------
template <bool RES, bool W32, bool WBF>
__global__ __launch_bounds__(256)
void ln_kernel(const float* __restrict__ xin, const float* __restrict__ res,
               const float* __restrict__ g, const float* __restrict__ bta,
               float* __restrict__ y32, bf16* __restrict__ ybf) {
  const int row = blockIdx.x;
  const int tid = threadIdx.x;
  __shared__ float r0[8], r1[8];
  size_t base = (size_t)row * kD;
  float v0 = xin[base + tid];
  float v1 = xin[base + tid + 256];
  if (RES) { v0 += res[base + tid]; v1 += res[base + tid + 256]; }
  float sum = v0 + v1, sq = v0 * v0 + v1 * v1;
#pragma unroll
  for (int m = 1; m < 32; m <<= 1) { sum += __shfl_xor(sum, m); sq += __shfl_xor(sq, m); }
  if ((tid & 31) == 0) { r0[tid >> 5] = sum; r1[tid >> 5] = sq; }
  __syncthreads();
  float ts = 0.f, tq = 0.f;
#pragma unroll
  for (int w = 0; w < 8; ++w) { ts += r0[w]; tq += r1[w]; }
  float mean = ts * (1.f / kD);
  float var  = tq * (1.f / kD) - mean * mean;
  float rstd = rsqrtf(var + 1e-5f);
  float y0 = (v0 - mean) * rstd * g[tid]       + bta[tid];
  float y1 = (v1 - mean) * rstd * g[tid + 256] + bta[tid + 256];
  if (W32) { y32[base + tid] = y0; y32[base + tid + 256] = y1; }
  if (WBF) { ybf[base + tid] = f2bf(y0); ybf[base + tid + 256] = f2bf(y1); }
}

// ---------------------------------------------------------------------------
// Host orchestration
// ---------------------------------------------------------------------------
extern "C" void kernel_launch(void* const* d_in, const int* in_sizes, int n_in,
                              void* d_out, int out_size, void* d_ws, size_t ws_size,
                              hipStream_t stream) {
  (void)in_sizes; (void)n_in; (void)out_size; (void)ws_size;
  const int*   x    = (const int*)  d_in[0];
  const float* emb  = (const float*)d_in[1];
  const float* pe   = (const float*)d_in[2];
  const float* Wqkv = (const float*)d_in[3];
  const float* Wo   = (const float*)d_in[4];
  const float* bo   = (const float*)d_in[5];
  const float* g1   = (const float*)d_in[6];
  const float* b1   = (const float*)d_in[7];
  const float* g2   = (const float*)d_in[8];
  const float* b2   = (const float*)d_in[9];
  const float* W1   = (const float*)d_in[10];
  const float* bf1  = (const float*)d_in[11];
  const float* W2   = (const float*)d_in[12];
  const float* bf2  = (const float*)d_in[13];
  const float* gf   = (const float*)d_in[14];
  const float* bfp  = (const float*)d_in[15];
  const float* Wout = (const float*)d_in[16];
  const float* bfc  = (const float*)d_in[17];

  char* ws = (char*)d_ws;
  size_t off = 0;
  auto carve = [&](size_t bytes) -> char* {
    char* p = ws + off;
    off = (off + bytes + 255) & ~(size_t)255;
    return p;
  };

  // transposed bf16 weights: [N][K] for every GEMM
  bf16*  wqkv_t = (bf16*)carve((size_t)kL * kD * 3 * kD * 2);
  bf16*  wo_t   = (bf16*)carve((size_t)kL * kD * kD * 2);
  bf16*  w1_t   = (bf16*)carve((size_t)kL * kD * kF * 2);
  bf16*  w2_t   = (bf16*)carve((size_t)kL * kF * kD * 2);
  bf16*  wout_t = (bf16*)carve((size_t)kD * kV * 2);
  float* h32    = (float*)carve((size_t)kM * kD * 4);
  bf16*  hbf    = (bf16*) carve((size_t)kM * kD * 2);
  bf16*  qkvb   = (bf16*) carve((size_t)kM * 3 * kD * 2);
  bf16*  attob  = (bf16*) carve((size_t)kM * kD * 2);
  float* proj   = (float*)carve((size_t)kM * kD * 4);
  bf16*  ffb    = (bf16*) carve((size_t)kM * kF * 2);
  float* ff2    = (float*)carve((size_t)kM * kD * 4);
  bf16*  hfb    = (bf16*) carve((size_t)kM * kD * 2);

  auto cvtT = [&](const float* src, bf16* dst, int K, int N) {
    cvt_t_kernel<<<dim3(N / 32, K / 32), dim3(256), 0, stream>>>(src, dst, K, N);
  };
  for (int l = 0; l < kL; ++l) {
    cvtT(Wqkv + (size_t)l * kD * 3 * kD, wqkv_t + (size_t)l * 3 * kD * kD, kD, 3 * kD);
    cvtT(Wo   + (size_t)l * kD * kD,     wo_t   + (size_t)l * kD * kD,     kD, kD);
    cvtT(W1   + (size_t)l * kD * kF,     w1_t   + (size_t)l * kF * kD,     kD, kF);
    cvtT(W2   + (size_t)l * kF * kD,     w2_t   + (size_t)l * kD * kF,     kF, kD);
  }
  cvtT(Wout, wout_t, kD, kV);

  embed_kernel<<<dim3(kM * kD / 256), dim3(256), 0, stream>>>(x, emb, pe, h32, hbf);

  for (int l = 0; l < kL; ++l) {
    gemm_bf16_kernel<0, true><<<dim3(3 * kD / BN, kM / BM), dim3(256), 0, stream>>>(
        hbf, wqkv_t + (size_t)l * 3 * kD * kD, nullptr, qkvb, kM, 3 * kD, kD);
    attn_kernel<<<dim3(kS / 64, kB * kH), dim3(128), 0, stream>>>(qkvb, attob);
    gemm_bf16_kernel<1, false><<<dim3(kD / BN, kM / BM), dim3(256), 0, stream>>>(
        attob, wo_t + (size_t)l * kD * kD, bo + (size_t)l * kD, proj, kM, kD, kD);
    ln_kernel<true, true, true><<<dim3(kM), dim3(256), 0, stream>>>(
        proj, h32, g1 + (size_t)l * kD, b1 + (size_t)l * kD, h32, hbf);
    gemm_bf16_kernel<2, true><<<dim3(kF / BN, kM / BM), dim3(256), 0, stream>>>(
        hbf, w1_t + (size_t)l * kF * kD, bf1 + (size_t)l * kF, ffb, kM, kF, kD);
    gemm_bf16_kernel<1, false><<<dim3(kD / BN, kM / BM), dim3(256), 0, stream>>>(
        ffb, w2_t + (size_t)l * kD * kF, bf2 + (size_t)l * kD, ff2, kM, kD, kF);
    ln_kernel<true, true, true><<<dim3(kM), dim3(256), 0, stream>>>(
        ff2, h32, g2 + (size_t)l * kD, b2 + (size_t)l * kD, h32, hbf);
  }

  ln_kernel<false, false, true><<<dim3(kM), dim3(256), 0, stream>>>(
      h32, nullptr, gf, bfp, nullptr, hfb);
  gemm_bf16_kernel<1, false><<<dim3(kV / BN, kM / BM), dim3(256), 0, stream>>>(
      hfb, wout_t, bfc, d_out, kM, kV, kD);
}
